// PhaseAttentionLayer_76081050681693
// MI455X (gfx1250) — matrix operand order repaired and checked by reading, not compiled, not run on hardware
//
#include <hip/hip_runtime.h>
#include <hip/hip_bf16.h>

// Problem constants (match reference)
#define BB   2
#define SS   2048
#define DD   768
#define HH   12
#define DH   64
#define FFD  3072
#define PDIM 64
#define MTOT (BB*SS)   // 4096

typedef __bf16 bf16;
typedef __attribute__((ext_vector_type(16))) __bf16 v16bf;
typedef __attribute__((ext_vector_type(8)))  __bf16 bf16x8;
typedef __attribute__((ext_vector_type(8)))  float  v8f;
typedef __attribute__((ext_vector_type(8)))  float  f32x8;

__device__ __forceinline__ bf16 f2bf(float x) { return (bf16)x; }

// ---------------------------------------------------------------------------
// WMMA fragment loaders (CDNA5 wave32 layouts, CDNA5 ISA 7.12.2)
// A: 16x32 bf16, M=lane%16. lanes 0-15 hold K={0..7,16..23}, lanes 16-31 K={8..15,24..31}
//    -> two contiguous 8xbf16 (16B) loads per lane from row-major [M,K].
// B: 32x16 bf16, N=lane%16. lanes 0-15 hold K={0..7,16..23}, lanes 16-31 K={8..15,24..31}
//    -> two contiguous 8xbf16 loads per lane from row-major [N,K].
// C/D f32 16x16: lane L: N=L%16, VGPR v: M = v + 8*(L/16).
// ---------------------------------------------------------------------------
__device__ __forceinline__ v16bf frag_a_global(const bf16* A, int lda, int m0, int k0) {
    int lane = threadIdx.x & 31;
    int hi = lane >> 4;
    const bf16* p = A + (size_t)(m0 + (lane & 15)) * lda + (k0 + hi * 8);
    bf16x8 c0 = *(const bf16x8*)p;        // K = hi*8 + {0..7}
    bf16x8 c1 = *(const bf16x8*)(p + 16); // K = hi*8 + {16..23}
    v16bf r;
#pragma unroll
    for (int i = 0; i < 8; ++i) { r[i] = c0[i]; r[i + 8] = c1[i]; }
    return r;
}

__device__ __forceinline__ v16bf frag_b_global(const bf16* Bw, int ldb, int n0, int k0) {
    int lane = threadIdx.x & 31;
    int hi = lane >> 4;
    const bf16* p = Bw + (size_t)(n0 + (lane & 15)) * ldb + (k0 + hi * 8);
    bf16x8 c0 = *(const bf16x8*)p;        // K = hi*8 + {0..7}
    bf16x8 c1 = *(const bf16x8*)(p + 16); // K = hi*8 + {16..23}
    v16bf r;
#pragma unroll
    for (int i = 0; i < 8; ++i) { r[i] = c0[i]; r[i + 8] = c1[i]; }
    return r;
}

// A-fragment gathered from an f32 stripe in LDS (row stride SS), converted to bf16
__device__ __forceinline__ v16bf frag_a_lds_f32(const float* sc, int k0) {
    int lane = threadIdx.x & 31;
    int hi = lane >> 4;
    const float* p = sc + (size_t)(lane & 15) * SS + (k0 + hi * 8);
    f32x8 c0 = *(const f32x8*)p;
    f32x8 c1 = *(const f32x8*)(p + 16);
    v16bf r;
#pragma unroll
    for (int i = 0; i < 8; ++i) { r[i] = f2bf(c0[i]); r[i + 8] = f2bf(c1[i]); }
    return r;
}

__device__ __forceinline__ v8f wmma_bf16(v16bf a, v16bf b, v8f c) {
    return __builtin_amdgcn_wmma_f32_16x16x32_bf16(false, a, false, b, (short)0, c,
                                                   false, false);
}

// ---------------------------------------------------------------------------
// f32 -> bf16 weight conversion
// ---------------------------------------------------------------------------
__global__ __launch_bounds__(256) void pa_cvt_kernel(const float* __restrict__ src,
                                                     bf16* __restrict__ dst, int n) {
    int i = blockIdx.x * 256 + threadIdx.x;
    if (i < n) dst[i] = f2bf(src[i]);
}

// ---------------------------------------------------------------------------
// LayerNorm over D=768, one row per block, writes bf16
// ---------------------------------------------------------------------------
__global__ __launch_bounds__(256) void pa_ln_kernel(const float* __restrict__ X,
                                                    const float* __restrict__ w,
                                                    const float* __restrict__ b,
                                                    bf16* __restrict__ out) {
    int row = blockIdx.x;
    const float* x = X + (size_t)row * DD;
    float s = 0.f, s2 = 0.f;
    for (int i = threadIdx.x; i < DD; i += 256) { float v = x[i]; s += v; s2 += v * v; }
#pragma unroll
    for (int o = 16; o > 0; o >>= 1) { s += __shfl_down(s, o); s2 += __shfl_down(s2, o); }
    __shared__ float sA[8], sB[8], stats[2];
    if ((threadIdx.x & 31) == 0) { sA[threadIdx.x >> 5] = s; sB[threadIdx.x >> 5] = s2; }
    __syncthreads();
    if (threadIdx.x == 0) {
        float t = 0.f, t2 = 0.f;
        for (int i = 0; i < 8; ++i) { t += sA[i]; t2 += sB[i]; }
        float m = t * (1.f / DD);
        float v = t2 * (1.f / DD) - m * m;
        stats[0] = m; stats[1] = rsqrtf(v + 1e-5f);
    }
    __syncthreads();
    float mean = stats[0], rstd = stats[1];
    for (int i = threadIdx.x; i < DD; i += 256)
        out[(size_t)row * DD + i] = f2bf((x[i] - mean) * rstd * w[i] + b[i]);
}

// ---------------------------------------------------------------------------
// Tiled transpose: xn [B*S, D] bf16 -> xnT [B, D, S] bf16
// ---------------------------------------------------------------------------
__global__ __launch_bounds__(256) void pa_transpose_kernel(const bf16* __restrict__ Xn,
                                                           bf16* __restrict__ XnT) {
    __shared__ bf16 tile[64][65];
    int b = blockIdx.z, s0 = blockIdx.x * 64, d0 = blockIdx.y * 64;
    for (int i = threadIdx.x; i < 64 * 64; i += 256) {
        int r = i >> 6, c = i & 63;
        tile[r][c] = Xn[((size_t)b * SS + s0 + r) * DD + d0 + c];
    }
    __syncthreads();
    for (int i = threadIdx.x; i < 64 * 64; i += 256) {
        int r = i >> 6, c = i & 63;  // r: d-offset, c: s-offset
        XnT[((size_t)b * DD + d0 + r) * SS + s0 + c] = tile[c][r];
    }
}

// ---------------------------------------------------------------------------
// Generic WMMA GEMM: out[M,N] = A[M,K] * Bw[N,K]^T, epilogue per MODE.
// 256 threads = 8 waves (4M x 2N); wave tile = 32(M) x 16*NT(N).
// NT=4 -> WG tile 128x128, 8 WMMA per 6 fragment loads (B reused over 2 M-rows,
// A reused over 4 N-tiles) -- the compute-dense config for the big GEMMs.
// NT=2 -> WG tile 128x64 for the narrow phase GEMM (N=64).
// MODE 0: QK proj  (+bias, split into q (scaled 1/8) / k, bf16 out)
// MODE 1: phase    (tanh, bf16 out)
// MODE 2: FFN1     (+bias, exact GELU, bf16 out)
// MODE 3: FFN2     (+bias, +residual, f32 out)
// ---------------------------------------------------------------------------
template <int MODE, int NT>
__global__ __launch_bounds__(256) void pa_gemm_kernel(
    const bf16* __restrict__ A, const bf16* __restrict__ Bw,
    const float* __restrict__ bias, const float* __restrict__ res,
    float* __restrict__ outf, bf16* __restrict__ outb0, bf16* __restrict__ outb1,
    int M, int N, int K) {
    (void)M; (void)N;
    int w = threadIdx.x >> 5, lane = threadIdx.x & 31;
    int m0 = blockIdx.y * 128 + (w >> 1) * 32;
    int n0 = blockIdx.x * (32 * NT) + (w & 1) * (16 * NT);
    v8f acc[2][NT] = {};
    for (int k0 = 0; k0 < K; k0 += 32) {
        if (k0 + 32 < K) {  // gfx1250 global_prefetch_b8 hints for next K-slab
            __builtin_prefetch(A + (size_t)m0 * K + (k0 + 32), 0, 3);
            __builtin_prefetch(Bw + (size_t)n0 * K + (k0 + 32), 0, 3);
        }
        v16bf a0 = frag_a_global(A, K, m0, k0);
        v16bf a1 = frag_a_global(A, K, m0 + 16, k0);
        v16bf bfr[NT];
#pragma unroll
        for (int j = 0; j < NT; ++j) bfr[j] = frag_b_global(Bw, K, n0 + 16 * j, k0);
#pragma unroll
        for (int j = 0; j < NT; ++j) {
            acc[0][j] = wmma_bf16(a0, bfr[j], acc[0][j]);
            acc[1][j] = wmma_bf16(a1, bfr[j], acc[1][j]);
        }
    }
    int hi = lane >> 4, nl = lane & 15;
#pragma unroll
    for (int i = 0; i < 2; ++i)
#pragma unroll
        for (int j = 0; j < NT; ++j)
#pragma unroll
            for (int v = 0; v < 8; ++v) {
                int m = m0 + i * 16 + 8 * hi + v;
                int n = n0 + j * 16 + nl;
                float val = acc[i][j][v];
                if constexpr (MODE == 0) {
                    val += bias[n];
                    if (n < DD) outb0[(size_t)m * DD + n] = f2bf(val * 0.125f);   // q / sqrt(64)
                    else        outb1[(size_t)m * DD + (n - DD)] = f2bf(val);     // k
                } else if constexpr (MODE == 1) {
                    outb0[(size_t)m * PDIM + n] = f2bf(tanhf(val));
                } else if constexpr (MODE == 2) {
                    val += bias[n];
                    val = 0.5f * val * (1.f + erff(val * 0.70710678118654752f));  // exact GELU
                    outb0[(size_t)m * FFD + n] = f2bf(val);
                } else {  // MODE == 3
                    val += bias[n];
                    outf[(size_t)m * DD + n] = res[(size_t)m * DD + n] + val;
                }
            }
}

// ---------------------------------------------------------------------------
// Head-averaged attention weights:
//   attn_w[b,q,k] = (1/H) * sum_h softmax_k( Q_h[q] . K_h[k] )   (scale folded into q)
// One WG per (b, 16-row q-tile); loops over heads; 16x2048 f32 score stripe in
// dynamic LDS (128 KB < 320 KB/WGP). Wave w owns columns [w*256, w*256+256).
// ---------------------------------------------------------------------------
__global__ __launch_bounds__(256) void pa_attn_kernel(const bf16* __restrict__ Q,
                                                      const bf16* __restrict__ Kk,
                                                      float* __restrict__ AW) {
    extern __shared__ float sc[];  // 16 * SS floats
    __shared__ float red[16][17];
    __shared__ float rowm[16], rows[16];
    int b = blockIdx.y, q0 = blockIdx.x * 16;
    int w = threadIdx.x >> 5, lane = threadIdx.x & 31;
    int hi = lane >> 4, nl = lane & 15;
    int r = threadIdx.x >> 4, sub = threadIdx.x & 15;
    const bf16* Qb = Q + (size_t)b * SS * DD;
    const bf16* Kb = Kk + (size_t)b * SS * DD;

    for (int h = 0; h < HH; ++h) {
        int kd = h * DH;
        v16bf a0 = frag_a_global(Qb, DD, q0, kd);
        v16bf a1 = frag_a_global(Qb, DD, q0, kd + 32);
        for (int t = 0; t < 16; ++t) {
            int n0 = w * 256 + t * 16;
            v16bf b0 = frag_b_global(Kb, DD, n0, kd);
            v16bf b1 = frag_b_global(Kb, DD, n0, kd + 32);
            v8f acc = {};
            acc = wmma_bf16(a0, b0, acc);
            acc = wmma_bf16(a1, b1, acc);
#pragma unroll
            for (int v = 0; v < 8; ++v)
                sc[(size_t)(8 * hi + v) * SS + n0 + nl] = acc[v];
        }
        __syncthreads();
        // row softmax (16 threads per row)
        float mx = -1e30f;
        for (int c = sub; c < SS; c += 16) mx = fmaxf(mx, sc[(size_t)r * SS + c]);
        red[r][sub] = mx;
        __syncthreads();
        if (sub == 0) {
            float m2 = red[r][0];
            for (int i = 1; i < 16; ++i) m2 = fmaxf(m2, red[r][i]);
            rowm[r] = m2;
        }
        __syncthreads();
        mx = rowm[r];
        float sm = 0.f;
        for (int c = sub; c < SS; c += 16) {
            float e = __expf(sc[(size_t)r * SS + c] - mx);
            sc[(size_t)r * SS + c] = e;
            sm += e;
        }
        red[r][sub] = sm;
        __syncthreads();
        if (sub == 0) {
            float t2 = 0.f;
            for (int i = 0; i < 16; ++i) t2 += red[r][i];
            rows[r] = t2;
        }
        __syncthreads();
        float inv = 1.f / (rows[r] * (float)HH);
        size_t base = ((size_t)b * SS + q0 + r) * SS;
        if (h == 0) for (int c = sub; c < SS; c += 16) AW[base + c] = sc[(size_t)r * SS + c] * inv;
        else        for (int c = sub; c < SS; c += 16) AW[base + c] += sc[(size_t)r * SS + c] * inv;
        __syncthreads();  // stripe reused next head
    }
}

// ---------------------------------------------------------------------------
// Blended softmax + context:
//   logits = log(attn_w + 1e-6) + alpha*0.5*(phase.phase^T + 1)      (WMMA, K=64)
//   p = softmax(logits);  xres = x + p @ xn                          (WMMA, K=2048)
// Same (b, 16-row) tiling, 128 KB LDS stripe. Wave w owns d-cols [w*96, w*96+96).
// ---------------------------------------------------------------------------
__global__ __launch_bounds__(256) void pa_blend_ctx_kernel(
    const bf16* __restrict__ P, const float* __restrict__ AW,
    const bf16* __restrict__ XnT, const float* __restrict__ X,
    const float* __restrict__ alphap, float* __restrict__ XR) {
    extern __shared__ float sc[];  // 16 * SS floats
    __shared__ float red[16][17];
    __shared__ float rowm[16], rowsum[16];
    int b = blockIdx.y, q0 = blockIdx.x * 16;
    int w = threadIdx.x >> 5, lane = threadIdx.x & 31;
    int hi = lane >> 4, nl = lane & 15;
    float alpha = alphap[0];
    const bf16* Pb = P + (size_t)b * SS * PDIM;

    v16bf pa0 = frag_a_global(Pb, PDIM, q0, 0);
    v16bf pa1 = frag_a_global(Pb, PDIM, q0, 32);
    for (int t = 0; t < 16; ++t) {
        int n0 = w * 256 + t * 16;
        v16bf pb0 = frag_b_global(Pb, PDIM, n0, 0);
        v16bf pb1 = frag_b_global(Pb, PDIM, n0, 32);
        v8f acc = {};
        acc = wmma_bf16(pa0, pb0, acc);
        acc = wmma_bf16(pa1, pb1, acc);
#pragma unroll
        for (int v = 0; v < 8; ++v) {
            int m = 8 * hi + v, n = n0 + nl;
            float aw = AW[((size_t)b * SS + q0 + m) * SS + n];
            sc[(size_t)m * SS + n] = __logf(aw + 1e-6f) + alpha * (0.5f * acc[v] + 0.5f);
        }
    }
    __syncthreads();

    int r = threadIdx.x >> 4, sub = threadIdx.x & 15;
    float mx = -1e30f;
    for (int c = sub; c < SS; c += 16) mx = fmaxf(mx, sc[(size_t)r * SS + c]);
    red[r][sub] = mx;
    __syncthreads();
    if (sub == 0) {
        float m2 = red[r][0];
        for (int i = 1; i < 16; ++i) m2 = fmaxf(m2, red[r][i]);
        rowm[r] = m2;
    }
    __syncthreads();
    mx = rowm[r];
    float sm = 0.f;
    for (int c = sub; c < SS; c += 16) {
        float e = __expf(sc[(size_t)r * SS + c] - mx);
        sc[(size_t)r * SS + c] = e;  // unnormalized prob; 1/sum folded into epilogue
        sm += e;
    }
    red[r][sub] = sm;
    __syncthreads();
    if (sub == 0) {
        float t2 = 0.f;
        for (int i = 0; i < 16; ++i) t2 += red[r][i];
        rowsum[r] = t2;
    }
    __syncthreads();

    // ctx GEMM: [16 x 2048] (LDS probs) @ xn[b] via XnT [D,S];  N-stripe per wave = 96
    const bf16* Bt = XnT + (size_t)b * DD * SS;
    v8f acc2[6] = {};
    for (int k0 = 0; k0 < SS; k0 += 32) {
        v16bf a = frag_a_lds_f32(sc, k0);
#pragma unroll
        for (int j = 0; j < 6; ++j) {
            v16bf bb = frag_b_global(Bt, SS, w * 96 + j * 16, k0);
            acc2[j] = wmma_bf16(a, bb, acc2[j]);
        }
    }
#pragma unroll
    for (int j = 0; j < 6; ++j)
#pragma unroll
        for (int v = 0; v < 8; ++v) {
            int m = 8 * hi + v;
            int n = w * 96 + j * 16 + nl;
            size_t gi = ((size_t)b * SS + q0 + m) * DD + n;
            XR[gi] = X[gi] + acc2[j][v] * (1.f / rowsum[m]);
        }
}

// ---------------------------------------------------------------------------
// Host-side launcher
// ---------------------------------------------------------------------------
extern "C" void kernel_launch(void* const* d_in, const int* in_sizes, int n_in,
                              void* d_out, int out_size, void* d_ws, size_t ws_size,
                              hipStream_t stream) {
    (void)in_sizes; (void)n_in; (void)out_size; (void)ws_size;
    const float* x      = (const float*)d_in[0];
    const float* ln1_w  = (const float*)d_in[1];
    const float* ln1_b  = (const float*)d_in[2];
    const float* ipw    = (const float*)d_in[3];   // [2304,768]
    const float* ipb    = (const float*)d_in[4];
    const float* phw    = (const float*)d_in[5];   // [64,768]
    const float* alphap = (const float*)d_in[6];
    const float* fw1    = (const float*)d_in[7];   // [3072,768]
    const float* fb1    = (const float*)d_in[8];
    const float* fw2    = (const float*)d_in[9];   // [768,3072]
    const float* fb2    = (const float*)d_in[10];
    const float* ln2_w  = (const float*)d_in[11];
    const float* ln2_b  = (const float*)d_in[12];
    float* out = (float*)d_out;

    // Workspace carve-up (~115 MB)
    uint8_t* p = (uint8_t*)d_ws;
    auto take = [&](size_t bytes) { uint8_t* r = p; p += (bytes + 255) & ~size_t(255); return r; };
    bf16*  xn_bf   = (bf16*)take((size_t)MTOT * DD * 2);
    bf16*  xnT_bf  = (bf16*)take((size_t)BB * DD * SS * 2);
    bf16*  q_bf    = (bf16*)take((size_t)MTOT * DD * 2);
    bf16*  k_bf    = (bf16*)take((size_t)MTOT * DD * 2);
    bf16*  ph_bf   = (bf16*)take((size_t)MTOT * PDIM * 2);
    bf16*  wqk_bf  = (bf16*)take((size_t)2 * DD * DD * 2);
    bf16*  wph_bf  = (bf16*)take((size_t)PDIM * DD * 2);
    bf16*  wf1_bf  = (bf16*)take((size_t)FFD * DD * 2);
    bf16*  wf2_bf  = (bf16*)take((size_t)DD * FFD * 2);
    float* attn_w  = (float*)take((size_t)BB * SS * SS * 4);
    float* xres    = (float*)take((size_t)MTOT * DD * 4);
    bf16*  x2n_bf  = (bf16*)take((size_t)MTOT * DD * 2);
    bf16*  h_bf    = (bf16*)take((size_t)MTOT * FFD * 2);

    auto cvt = [&](const float* s, bf16* d, int n) {
        pa_cvt_kernel<<<(n + 255) / 256, 256, 0, stream>>>(s, d, n);
    };
    cvt(ipw, wqk_bf, 2 * DD * DD);      // q,k rows only (first 1536 of in_proj_w)
    cvt(phw, wph_bf, PDIM * DD);
    cvt(fw1, wf1_bf, FFD * DD);
    cvt(fw2, wf2_bf, DD * FFD);

    // LN1 -> xn (bf16), and transposed copy for context GEMM B-operand
    pa_ln_kernel<<<MTOT, 256, 0, stream>>>(x, ln1_w, ln1_b, xn_bf);
    pa_transpose_kernel<<<dim3(SS / 64, DD / 64, BB), 256, 0, stream>>>(xn_bf, xnT_bf);

    // Q/K projection: [4096,1536] = xn @ Wqk^T + b ; q pre-scaled by 1/sqrt(Dh)
    pa_gemm_kernel<0, 4><<<dim3((2 * DD) / 128, MTOT / 128), 256, 0, stream>>>(
        xn_bf, wqk_bf, ipb, nullptr, nullptr, q_bf, k_bf, MTOT, 2 * DD, DD);

    // phase = tanh(xn @ phase_w^T): [4096,64]
    pa_gemm_kernel<1, 2><<<dim3(PDIM / 64, MTOT / 128), 256, 0, stream>>>(
        xn_bf, wph_bf, nullptr, nullptr, nullptr, ph_bf, nullptr, MTOT, PDIM, DD);

    // head-averaged softmax(QK^T)
    pa_attn_kernel<<<dim3(SS / 16, BB), 256, 16 * SS * sizeof(float), stream>>>(
        q_bf, k_bf, attn_w);

    // blended softmax + context + residual -> xres
    pa_blend_ctx_kernel<<<dim3(SS / 16, BB), 256, 16 * SS * sizeof(float), stream>>>(
        ph_bf, attn_w, xnT_bf, x, alphap, xres);

    // LN2
    pa_ln_kernel<<<MTOT, 256, 0, stream>>>(xres, ln2_w, ln2_b, x2n_bf);

    // FFN
    pa_gemm_kernel<2, 4><<<dim3(FFD / 128, MTOT / 128), 256, 0, stream>>>(
        x2n_bf, wf1_bf, fb1, nullptr, nullptr, h_bf, nullptr, MTOT, FFD, DD);
    pa_gemm_kernel<3, 4><<<dim3(DD / 128, MTOT / 128), 256, 0, stream>>>(
        h_bf, wf2_bf, fb2, xres, out, nullptr, nullptr, MTOT, DD, FFD);
}
